// HeteroGNN_9225589752381
// MI455X (gfx1250) — matrix operand order repaired and checked by reading, not compiled, not run on hardware
//
#include <hip/hip_runtime.h>
#include <stdint.h>

// HeteroGNN on MI455X (gfx1250): bf16 WMMA GEMMs + f32 global-atomic scatter-mean.
// All dense matmuls go through v_wmma_f32_16x16x32_bf16 (16x16 tiles, wave32).
// Workspace requirement: ~210 MB (see layout in kernel_launch).

#define TPB 256

typedef float        v8f   __attribute__((ext_vector_type(8)));
typedef unsigned int v8u   __attribute__((ext_vector_type(8)));
typedef __bf16       v16bf __attribute__((ext_vector_type(16)));

__device__ __forceinline__ unsigned short f2bf(float f) {
  unsigned int u = __builtin_bit_cast(unsigned int, f);
  u += 0x7FFFu + ((u >> 16) & 1u);          // round-to-nearest-even
  return (unsigned short)(u >> 16);
}

// ---------------------------------------------------------------- utility
__global__ void k_zero(float* __restrict__ p, long long n) {
  long long i = (long long)blockIdx.x * TPB + threadIdx.x;
  if (i < n) p[i] = 0.0f;
}

__global__ void k_f32_to_bf16(const float* __restrict__ x,
                              unsigned short* __restrict__ y, long long n) {
  long long i = (long long)blockIdx.x * TPB + threadIdx.x;
  if (i < n) y[i] = f2bf(x[i]);
}

// ---------------------------------------------------------------- scatter-mean
__global__ void k_count(const int* __restrict__ di, float* __restrict__ cnt, int E) {
  int i = blockIdx.x * TPB + threadIdx.x;
  if (i < E) unsafeAtomicAdd(&cnt[di[i]], 1.0f);   // global_atomic_add_f32
}

__global__ void k_scatter(const float* __restrict__ xs, const int* __restrict__ si,
                          const int* __restrict__ di, float* __restrict__ agg,
                          long long total, int kshift, int kmask) {
  long long i = (long long)blockIdx.x * TPB + threadIdx.x;
  if (i >= total) return;
  int e = (int)(i >> kshift);
  int f = (int)i & kmask;
  unsafeAtomicAdd(&agg[((size_t)di[e] << kshift) + f],
                  xs[((size_t)si[e] << kshift) + f]);
}

__global__ void k_mean_bf16(const float* __restrict__ agg, const float* __restrict__ cnt,
                            unsigned short* __restrict__ out, long long total, int kshift) {
  long long i = (long long)blockIdx.x * TPB + threadIdx.x;
  if (i >= total) return;
  int n   = (int)(i >> kshift);
  float c = cnt[n];
  float d = c > 1.0f ? c : 1.0f;
  out[i] = f2bf(agg[i] / d);
}

// ---------------------------------------------------------------- weight repack
// Pack W[K x 128] (f32, row-major) into per-lane-contiguous WMMA B fragments:
// pk[(kb*8 + ntile)*256 + lane*8 + v] = {bf16(W[k0+1,n]), bf16(W[k0,n])}
// with k0 = kb*32 + (lane>=16 ? 8 : 0) + (v>=4 ? 16 : 0) + 2*(v&3), n = ntile*16 + (lane&15).
__global__ void k_pack_w(const float* __restrict__ W, int K, unsigned int* __restrict__ pk) {
  int idx = blockIdx.x * TPB + threadIdx.x;
  int total = (K >> 5) * 2048;
  if (idx >= total) return;
  int v  = idx & 7;
  int L  = (idx >> 3) & 31;
  int nt = (idx >> 8) & 7;
  int kb = idx >> 11;
  int k0 = kb * 32 + ((L & 16) ? 8 : 0) + ((v & 4) ? 16 : 0) + 2 * (v & 3);
  int n  = nt * 16 + (L & 15);
  unsigned short lo = f2bf(W[(size_t)k0 * 128 + n]);
  unsigned short hi = f2bf(W[(size_t)(k0 + 1) * 128 + n]);
  pk[idx] = ((unsigned int)hi << 16) | lo;
}

// ---------------------------------------------------------------- WMMA GEMM
// C[N x 128] = alpha * (A[N x K](bf16) @ Bpk + bias) (+ C if accum)
// Block: 256 threads = 8 waves; block computes a 16x128 stripe, wave w -> cols [16w,16w+16).
__global__ __launch_bounds__(TPB) void k_gemm_bf16(
    const unsigned short* __restrict__ A, int N, int K,
    const unsigned int* __restrict__ Bpk, const float* __restrict__ bias,
    float* Cf, unsigned short* Cbf, int accum, float alpha) {
  __shared__ unsigned short smA[16 * 264];   // 16 rows, stride K+8 (<= 264)

  const int tid     = threadIdx.x;
  const int m0      = blockIdx.x * 16;
  const int strideE = K + 8;
  const int rowDw   = K >> 1;

  // cooperative, coalesced A-tile load (zero-padded past N)
  for (int i = tid; i < 16 * rowDw; i += TPB) {
    int row = i / rowDw;
    int cd  = i - row * rowDw;
    unsigned int val = 0u;
    int m = m0 + row;
    if (m < N) val = *(const unsigned int*)(A + (size_t)m * K + 2 * cd);
    *(unsigned int*)(smA + row * strideE + 2 * cd) = val;
  }
  __syncthreads();

  const int lane  = tid & 31;
  const int wv    = tid >> 5;                 // n-tile 0..7
  const int m16   = lane & 15;
  const int kbase = (lane < 16) ? 0 : 8;      // 16-bit A-fragment lane layout
  const unsigned short* aRow = smA + m16 * strideE;

  __builtin_prefetch(Bpk + (size_t)wv * 256 + lane * 8, 0, 3);

  v8f c = {0.f, 0.f, 0.f, 0.f, 0.f, 0.f, 0.f, 0.f};
  const int nkb = K >> 5;
  for (int kb = 0; kb < nkb; ++kb) {
    const uint4 lo4 = *(const uint4*)(aRow + kb * 32 + kbase);        // K = kb*32+kbase..+7
    const uint4 hi4 = *(const uint4*)(aRow + kb * 32 + 16 + kbase);   // K = +16..+23
    v8u au = {lo4.x, lo4.y, lo4.z, lo4.w, hi4.x, hi4.y, hi4.z, hi4.w};
    v8u bu = *(const v8u*)(Bpk + ((size_t)(kb * 8 + wv) << 8) + lane * 8);
    c = __builtin_amdgcn_wmma_f32_16x16x32_bf16(
        false, __builtin_bit_cast(v16bf, au),
        false, __builtin_bit_cast(v16bf, bu),
        (short)0, c, false, false);
  }

  const int   n   = wv * 16 + m16;            // C/D layout: N = lane&15
  const float bv  = bias ? bias[n] : 0.0f;
  const int   mhi = (lane < 16) ? 0 : 8;      // C/D layout: M = r (+8 for upper lanes)
#pragma unroll
  for (int r = 0; r < 8; ++r) {
    int m = m0 + r + mhi;
    if (m < N) {
      float acc = (c[r] + bv) * alpha;
      size_t off = (size_t)m * 128 + n;
      if (accum) acc += Cf[off];
      if (Cf)  Cf[off]  = acc;
      if (Cbf) Cbf[off] = f2bf(acc);
    }
  }
}

// ---------------------------------------------------------------- BatchNorm
__global__ void k_colsum(const float* __restrict__ X, int N,
                         float* __restrict__ sums, float* __restrict__ sumsq) {
  int c  = threadIdx.x;                // 128 threads = 128 columns
  int r0 = blockIdx.x * 512;
  int r1 = r0 + 512; if (r1 > N) r1 = N;
  float s = 0.f, q = 0.f;
  for (int r = r0; r < r1; ++r) {
    float v = X[(size_t)r * 128 + c];
    s += v; q += v * v;
  }
  unsafeAtomicAdd(&sums[c], s);
  unsafeAtomicAdd(&sumsq[c], q);
}

__global__ void k_bn_act(const float* X, const float* __restrict__ sums,
                         const float* __restrict__ sumsq,
                         const float* __restrict__ gamma, const float* __restrict__ beta,
                         long long total, float invN, int leaky,
                         unsigned short* outbf, float* outf) {
  long long i = (long long)blockIdx.x * TPB + threadIdx.x;
  if (i >= total) return;
  int c = (int)i & 127;
  float mu  = sums[c] * invN;
  float var = sumsq[c] * invN - mu * mu;     // biased variance (training-mode BN)
  float y = (X[i] - mu) * rsqrtf(var + 1e-5f) * gamma[c] + beta[c];
  if (leaky && y < 0.f) y *= 0.01f;
  if (outf)  outf[i]  = y;
  if (outbf) outbf[i] = f2bf(y);
}

// ---------------------------------------------------------------- dot head
__global__ __launch_bounds__(TPB) void k_dot(const float* __restrict__ Xa,
                                             const float* __restrict__ Xb,
                                             const int* __restrict__ ia,
                                             const int* __restrict__ ib,
                                             float* __restrict__ out, int L) {
  int e    = blockIdx.x * 8 + (threadIdx.x >> 5);   // one wave32 per edge
  int lane = threadIdx.x & 31;
  if (e >= L) return;
  const float4 a = *(const float4*)(Xa + (size_t)ia[e] * 128 + lane * 4);
  const float4 b = *(const float4*)(Xb + (size_t)ib[e] * 128 + lane * 4);
  float s = a.x * b.x + a.y * b.y + a.z * b.z + a.w * b.w;
  for (int o = 16; o > 0; o >>= 1) s += __shfl_xor(s, o, 32);
  if (lane == 0) out[e] = s;
}

// ---------------------------------------------------------------- host side
static inline unsigned int divup_u(long long a, long long b) {
  return (unsigned int)((a + b - 1) / b);
}

struct RelW { const float *Wd, *bd, *Ws, *bs, *Wu, *bu; };

static void run_conv(hipStream_t s,
                     const float* xsrc, int Ksrc,
                     const unsigned short* adst_bf, int Ndst, int Kdst,
                     const int* si, const int* di, int E,
                     const RelW& w, float* ebuf, int accumFirst, float alpha,
                     float* agg, float* cnt, unsigned int* pk,
                     unsigned short* aggbf, unsigned short* t1, unsigned short* t2) {
  long long aggN = (long long)Ndst * Ksrc;
  k_zero<<<divup_u(aggN, TPB), TPB, 0, s>>>(agg, aggN);
  k_zero<<<divup_u(Ndst, TPB), TPB, 0, s>>>(cnt, Ndst);
  k_count<<<divup_u(E, TPB), TPB, 0, s>>>(di, cnt, E);
  int kshift = (Ksrc == 128) ? 7 : 6;
  long long tot = (long long)E * Ksrc;
  k_scatter<<<divup_u(tot, TPB), TPB, 0, s>>>(xsrc, si, di, agg, tot, kshift, Ksrc - 1);
  k_mean_bf16<<<divup_u(aggN, TPB), TPB, 0, s>>>(agg, cnt, aggbf, aggN, kshift);

  unsigned int gB = divup_u(Ndst, 16);
  // dst_msg = x_dst @ Wd + bd  -> t1 (bf16)
  k_pack_w<<<divup_u((Kdst >> 5) * 2048, TPB), TPB, 0, s>>>(w.Wd, Kdst, pk);
  k_gemm_bf16<<<gB, TPB, 0, s>>>(adst_bf, Ndst, Kdst, pk, w.bd, nullptr, t1, 0, 1.0f);
  // src_msg = agg @ Ws + bs  -> t2 (bf16)
  k_pack_w<<<divup_u((Ksrc >> 5) * 2048, TPB), TPB, 0, s>>>(w.Ws, Ksrc, pk);
  k_gemm_bf16<<<gB, TPB, 0, s>>>(aggbf, Ndst, Ksrc, pk, w.bs, nullptr, t2, 0, 1.0f);
  // e = alpha*(t1 @ Wu[0:128] + t2 @ Wu[128:256] + bu)  (concat folded into two GEMMs)
  k_pack_w<<<divup_u(8 * 2048, TPB), TPB, 0, s>>>(w.Wu, 128, pk);
  k_gemm_bf16<<<gB, TPB, 0, s>>>(t1, Ndst, 128, pk, w.bu, ebuf, nullptr, accumFirst, alpha);
  k_pack_w<<<divup_u(8 * 2048, TPB), TPB, 0, s>>>(w.Wu + 128 * 128, 128, pk);
  k_gemm_bf16<<<gB, TPB, 0, s>>>(t2, Ndst, 128, pk, nullptr, ebuf, nullptr, 1, alpha);
}

static void run_bn(hipStream_t s, float* X, int N, const float* gamma, const float* beta,
                   int leaky, float* stats, unsigned short* outbf, float* outf) {
  k_zero<<<1, TPB, 0, s>>>(stats, 256);
  k_colsum<<<divup_u(N, 512), 128, 0, s>>>(X, N, stats, stats + 128);
  long long tot = (long long)N * 128;
  k_bn_act<<<divup_u(tot, TPB), TPB, 0, s>>>(X, stats, stats + 128, gamma, beta,
                                             tot, 1.0f / (float)N, leaky, outbf, outf);
}

extern "C" void kernel_launch(void* const* d_in, const int* in_sizes, int n_in,
                              void* d_out, int out_size, void* d_ws, size_t ws_size,
                              hipStream_t stream) {
  (void)n_in; (void)out_size; (void)ws_size;
  // inputs 0..11 in setup_inputs() insertion order
  const float* xg = (const float*)d_in[0];
  const float* xd = (const float*)d_in[1];
  const int* ppi_src = (const int*)d_in[2];
  const int* ppi_dst = (const int*)d_in[3];
  const int* g2d_src = (const int*)d_in[4];
  const int* g2d_dst = (const int*)d_in[5];
  const int* d2g_src = (const int*)d_in[6];
  const int* d2g_dst = (const int*)d_in[7];
  const int* lab_g2d_src = (const int*)d_in[8];
  const int* lab_g2d_dst = (const int*)d_in[9];
  const int* lab_d2g_src = (const int*)d_in[10];
  const int* lab_d2g_dst = (const int*)d_in[11];
  // params: jax tree-flatten order (dict keys sorted, DFS):
  // bn1{dis{b,g},gene{b,g}}, bn2{...}, conv1{d2g,g2d,ppi}x{dst,src,upd}x{W,b}, conv2{...}
  const float* bn1_dis_b  = (const float*)d_in[12];
  const float* bn1_dis_g  = (const float*)d_in[13];
  const float* bn1_gene_b = (const float*)d_in[14];
  const float* bn1_gene_g = (const float*)d_in[15];
  const float* bn2_dis_b  = (const float*)d_in[16];
  const float* bn2_dis_g  = (const float*)d_in[17];
  const float* bn2_gene_b = (const float*)d_in[18];
  const float* bn2_gene_g = (const float*)d_in[19];
  auto rel = [&](int base) -> RelW {
    return { (const float*)d_in[base + 0], (const float*)d_in[base + 1],
             (const float*)d_in[base + 2], (const float*)d_in[base + 3],
             (const float*)d_in[base + 4], (const float*)d_in[base + 5] };
  };
  RelW c1_d2g = rel(20), c1_g2d = rel(26), c1_ppi = rel(32);
  RelW c2_d2g = rel(38), c2_g2d = rel(44), c2_ppi = rel(50);

  const int FG = 128, FD = 64;
  const int NGn = in_sizes[0] / FG;
  const int NDn = in_sizes[1] / FD;
  const int EP  = in_sizes[2];
  const int EG  = in_sizes[4];
  const int Ln  = in_sizes[8];

  // workspace layout (~210 MB)
  char* p = (char*)d_ws;
  auto alloc = [&](size_t bytes) -> char* {
    char* r = p; p += (bytes + 255) & ~(size_t)255; return r;
  };
  unsigned int*   pk      = (unsigned int*)  alloc(64 * 1024);
  float*          cnt     = (float*)         alloc((size_t)NGn * 4);
  float*          stats   = (float*)         alloc(256 * 4);
  float*          agg     = (float*)         alloc((size_t)NGn * 128 * 4);
  float*          gbuf    = (float*)         alloc((size_t)NGn * 128 * 4);
  float*          dbuf    = (float*)         alloc((size_t)NDn * 128 * 4);
  float*          gactf   = (float*)         alloc((size_t)NGn * 128 * 4);
  float*          dactf   = (float*)         alloc((size_t)NDn * 128 * 4);
  unsigned short* xg_bf   = (unsigned short*)alloc((size_t)NGn * 128 * 2);
  unsigned short* xd_bf   = (unsigned short*)alloc((size_t)NDn * 64 * 2);
  unsigned short* aggbf   = (unsigned short*)alloc((size_t)NGn * 128 * 2);
  unsigned short* t1      = (unsigned short*)alloc((size_t)NGn * 128 * 2);
  unsigned short* t2      = (unsigned short*)alloc((size_t)NGn * 128 * 2);
  unsigned short* gact_bf = (unsigned short*)alloc((size_t)NGn * 128 * 2);
  unsigned short* dact_bf = (unsigned short*)alloc((size_t)NDn * 128 * 2);

  // bf16 copies of the inputs (WMMA A operands)
  k_f32_to_bf16<<<divup_u((long long)NGn * FG, TPB), TPB, 0, stream>>>(xg, xg_bf, (long long)NGn * FG);
  k_f32_to_bf16<<<divup_u((long long)NDn * FD, TPB), TPB, 0, stream>>>(xd, xd_bf, (long long)NDn * FD);

  // ---- layer 1 ----  gene = 0.5*(e_ppi + e_d2g), disease = e_g2d
  run_conv(stream, xg, FG, xg_bf, NGn, FG, ppi_src, ppi_dst, EP, c1_ppi, gbuf, 0, 0.5f,
           agg, cnt, pk, aggbf, t1, t2);
  run_conv(stream, xd, FD, xg_bf, NGn, FG, d2g_src, d2g_dst, EG, c1_d2g, gbuf, 1, 0.5f,
           agg, cnt, pk, aggbf, t1, t2);
  run_conv(stream, xg, FG, xd_bf, NDn, FD, g2d_src, g2d_dst, EG, c1_g2d, dbuf, 0, 1.0f,
           agg, cnt, pk, aggbf, t1, t2);
  run_bn(stream, gbuf, NGn, bn1_gene_g, bn1_gene_b, 1, stats, gact_bf, gactf);
  run_bn(stream, dbuf, NDn, bn1_dis_g,  bn1_dis_b,  1, stats, dact_bf, dactf);

  // ---- layer 2 ----
  run_conv(stream, gactf, 128, gact_bf, NGn, 128, ppi_src, ppi_dst, EP, c2_ppi, gbuf, 0, 0.5f,
           agg, cnt, pk, aggbf, t1, t2);
  run_conv(stream, dactf, 128, gact_bf, NGn, 128, d2g_src, d2g_dst, EG, c2_d2g, gbuf, 1, 0.5f,
           agg, cnt, pk, aggbf, t1, t2);
  run_conv(stream, gactf, 128, dact_bf, NDn, 128, g2d_src, g2d_dst, EG, c2_g2d, dbuf, 0, 1.0f,
           agg, cnt, pk, aggbf, t1, t2);
  run_bn(stream, gbuf, NGn, bn2_gene_g, bn2_gene_b, 0, stats, nullptr, gbuf);  // in-place f32
  run_bn(stream, dbuf, NDn, bn2_dis_g,  bn2_dis_b,  0, stats, nullptr, dbuf);

  // ---- dot-product head ----
  float* out = (float*)d_out;
  k_dot<<<divup_u(Ln, 8), TPB, 0, stream>>>(gbuf, dbuf, lab_g2d_src, lab_g2d_dst, out, Ln);
  k_dot<<<divup_u(Ln, 8), TPB, 0, stream>>>(dbuf, gbuf, lab_d2g_src, lab_d2g_dst, out + Ln, Ln);
}
